// GPT_67233418051824
// MI455X (gfx1250) — matrix operand order repaired and checked by reading, not compiled, not run on hardware
//
#include <hip/hip_runtime.h>
#include <hip/hip_bf16.h>

// ---------------------------------------------------------------------------
// GPT forward for MI455X (gfx1250): bf16 WMMA GEMMs fed by the Tensor Data
// Mover (async DMA -> LDS, double buffered), fused epilogues.
// ---------------------------------------------------------------------------

typedef __attribute__((ext_vector_type(16))) __bf16        v16bf;
typedef __attribute__((ext_vector_type(8)))  float         v8f;
typedef __attribute__((ext_vector_type(4)))  unsigned int  u32x4;
typedef __attribute__((ext_vector_type(8)))  int           i32x8;
typedef __attribute__((ext_vector_type(4)))  int           i32x4;

#ifndef __has_builtin
#define __has_builtin(x) 0
#endif
#if __has_builtin(__builtin_amdgcn_tensor_load_to_lds)
#define USE_TDM 1
#else
#define USE_TDM 0
#endif

union Frag {
    v16bf v;
    uint4 q[2];
};

__device__ __forceinline__ unsigned short f2bf(float a) {
    unsigned u = __float_as_uint(a);
    return (unsigned short)((u + 0x7FFFu + ((u >> 16) & 1u)) >> 16);  // RNE
}

__device__ __forceinline__ float gelu_exact(float x) {
    return 0.5f * x * (1.0f + erff(x * 0.70710678118654752f));
}

#if USE_TDM
// Issue one 2D TDM tile load: global (bf16, row stride = stride_elems) -> LDS,
// tile = tile_rows x tile_k elements, packed row-major (row stride tile_k) in LDS.
// D# per CDNA5 ISA 8.3/8.4 (group0: count/lds/global/type, group1: dims/strides).
// This toolchain's builtin takes 6 args: (g0, g1, g2, g3, g_extra, cpol).
__device__ __forceinline__ void tdm_load_2d(unsigned lds_off, const void* gaddr,
                                            unsigned tile_k, unsigned tile_rows,
                                            unsigned tensor_k, unsigned tensor_rows,
                                            unsigned long long stride_elems)
{
    const unsigned long long ga = (unsigned long long)gaddr;
    u32x4 g0;
    g0[0] = 1u;                                    // count=1, user mode
    g0[1] = lds_off;                               // LDS byte address
    g0[2] = (unsigned)ga;                          // global_addr[31:0]
    g0[3] = (unsigned)(ga >> 32) | (2u << 30);     // global_addr[56:32] | type=2
    i32x8 g1;
    g1[0] = (int)(1u << 16);                       // workgroup_mask=0, data_size=1 (2B)
    g1[1] = (int)((tensor_k & 0xFFFFu) << 16);     // tensor_dim0[15:0]
    g1[2] = (int)((tensor_k >> 16) | ((tensor_rows & 0xFFFFu) << 16));
    g1[3] = (int)((tensor_rows >> 16) | (tile_k << 16));   // tile_dim0
    g1[4] = (int)tile_rows;                        // tile_dim1 (tile_dim2 = 0)
    g1[5] = (int)(unsigned)(stride_elems & 0xFFFFFFFFull); // dim0_stride[31:0]
    g1[6] = (int)(unsigned)(stride_elems >> 32);           // dim0_stride[47:32]
    g1[7] = 0;
    const i32x4 z4 = {0, 0, 0, 0};
    const i32x8 z8 = {0, 0, 0, 0, 0, 0, 0, 0};
    __builtin_amdgcn_tensor_load_to_lds(g0, g1, z4, z4, z8, 0);
}
#endif

// ---------------------------------------------------------------------------
// GEMM: C[M,N] = epilogue( A[M,K] @ W[N,K]^T ), A/W bf16, f32 accumulate.
// epilogue: 0 = f32 store, 1 = exact GELU + bf16 store, 2 = f32 residual add.
// Block 256 threads (8 waves), block tile 128x128, wave tile 32x64, K-step 64.
// TDM double-buffers both tiles into LDS; 16 v_wmma per stage.
// Requires M%128==0, N%128==0, K%64==0 (holds for every call here).
// ---------------------------------------------------------------------------
#define BM 128
#define BN 128
#define KT 64

__global__ __launch_bounds__(256)
void gemm_bf16_tdm(const unsigned short* __restrict__ A,
                   const unsigned short* __restrict__ W,
                   const float* __restrict__ R,
                   float* __restrict__ Cf,
                   unsigned short* __restrict__ Cb,
                   int M, int N, int K, int epilogue)
{
    __shared__ __align__(16) unsigned short As[2][BM * KT];   // 32 KB
    __shared__ __align__(16) unsigned short Bs[2][BN * KT];   // 32 KB

    const int tid    = threadIdx.x;
    const int wave   = tid >> 5;
    const int lane   = tid & 31;
    const int lane16 = lane & 15;
    const int khalf  = lane >> 4;
    const int kbA    = khalf << 3;        // A frag K base: 0 / 8
    const int kbB    = khalf << 3;        // B frag K base: 0 / 8

    const int m0 = blockIdx.y * BM;
    const int n0 = blockIdx.x * BN;
    const int wr = wave & 3;              // 4 row groups of 32 rows
    const int wc = wave >> 2;             // 2 col groups of 64 cols

    v8f c[2][4] = {};
    const int nstage = K / KT;

#if USE_TDM
    auto issue = [&](int s) {
        const int k0 = s * KT;
        tdm_load_2d((unsigned)(size_t)&As[s & 1][0], A + (size_t)m0 * K + k0,
                    KT, BM, (unsigned)K, (unsigned)M, (unsigned long long)K);
        tdm_load_2d((unsigned)(size_t)&Bs[s & 1][0], W + (size_t)n0 * K + k0,
                    KT, BN, (unsigned)K, (unsigned)N, (unsigned long long)K);
    };
    if (wave == 0) issue(0);
#endif

    for (int s = 0; s < nstage; ++s) {
#if USE_TDM
        if (wave == 0) {
            if (s + 1 < nstage) {
                issue(s + 1);                                  // prefetch next stage
                __builtin_amdgcn_s_wait_tensorcnt(1);          // stage s resident
            } else {
                __builtin_amdgcn_s_wait_tensorcnt(0);
            }
        }
        __syncthreads();
        const unsigned short* as = &As[s & 1][0];
        const unsigned short* bs = &Bs[s & 1][0];
#else
        {   // fallback: cooperative vector staging (bf16 -> LDS), single buffer
            const int srow = tid >> 3, squad = tid & 7;        // 8 thr/row, 8 elems
            const int k0 = s * KT;
            #pragma unroll
            for (int p = 0; p < 4; ++p) {
                const int row = p * 32 + srow;
                *reinterpret_cast<uint4*>(&As[0][row * KT + squad * 8]) =
                    *reinterpret_cast<const uint4*>(&A[(size_t)(m0 + row) * K + k0 + squad * 8]);
                *reinterpret_cast<uint4*>(&Bs[0][row * KT + squad * 8]) =
                    *reinterpret_cast<const uint4*>(&W[(size_t)(n0 + row) * K + k0 + squad * 8]);
            }
            __syncthreads();
        }
        const unsigned short* as = &As[0][0];
        const unsigned short* bs = &Bs[0][0];
#endif

        #pragma unroll
        for (int kk = 0; kk < 2; ++kk) {                       // two 16x16x32 K-steps
            Frag a[2], b[4];
            #pragma unroll
            for (int mr = 0; mr < 2; ++mr) {
                const int row = wr * 32 + mr * 16 + lane16;
                a[mr].q[0] = *reinterpret_cast<const uint4*>(&as[row * KT + kk * 32 + kbA]);
                a[mr].q[1] = *reinterpret_cast<const uint4*>(&as[row * KT + kk * 32 + kbA + 16]);
            }
            #pragma unroll
            for (int nc = 0; nc < 4; ++nc) {
                const int col = wc * 64 + nc * 16 + lane16;
                b[nc].q[0] = *reinterpret_cast<const uint4*>(&bs[col * KT + kk * 32 + kbB]);
                b[nc].q[1] = *reinterpret_cast<const uint4*>(&bs[col * KT + kk * 32 + kbB + 16]);
            }
            #pragma unroll
            for (int mr = 0; mr < 2; ++mr)
                #pragma unroll
                for (int nc = 0; nc < 4; ++nc)
                    c[mr][nc] = __builtin_amdgcn_wmma_f32_16x16x32_bf16(
                        false, a[mr].v, false, b[nc].v,
                        (short)0, c[mr][nc], false, false);
        }
        __syncthreads();
    }

    // epilogue + store (D layout: N = lane16, M = r + 8*khalf)
    #pragma unroll
    for (int mr = 0; mr < 2; ++mr) {
        #pragma unroll
        for (int nc = 0; nc < 4; ++nc) {
            const int col     = n0 + wc * 64 + nc * 16 + lane16;
            const int rowbase = m0 + wr * 32 + mr * 16 + khalf * 8;
            #pragma unroll
            for (int r = 0; r < 8; ++r) {
                const size_t idx = (size_t)(rowbase + r) * N + col;
                const float v = c[mr][nc][r];
                if (epilogue == 1)      Cb[idx] = f2bf(gelu_exact(v));
                else if (epilogue == 2) Cf[idx] = v + R[idx];
                else                    Cf[idx] = v;
            }
        }
    }
}

// ---------------------------------------------------------------------------
// Causal attention, exact softmax. B=2, H=16, S=1024, D=1024, HD=64.
// One block = 8 waves = 8 query rows of one (b,h); output written as bf16.
// ---------------------------------------------------------------------------
__global__ __launch_bounds__(256)
void attention_kernel(const float* __restrict__ Q, const float* __restrict__ Kc,
                      const float* __restrict__ Vc, unsigned short* __restrict__ O)
{
    constexpr int S = 1024, D = 1024, HD = 64;
    constexpr float scale = 0.125f;

    __shared__ float sc[8][S];
    __shared__ float qb[8][HD];

    const int wave = threadIdx.x >> 5;
    const int lane = threadIdx.x & 31;
    const int qblk = blockIdx.x & 127;
    const int h    = (blockIdx.x >> 7) & 15;
    const int b    = blockIdx.x >> 11;

    const int qrow = qblk * 8 + wave;
    const size_t qoff = ((size_t)(b * S + qrow)) * D + h * HD;

    qb[wave][lane]      = Q[qoff + lane];
    qb[wave][lane + 32] = Q[qoff + lane + 32];

    float lmax = -INFINITY;
    for (int key = lane; key < S; key += 32) {
        float s;
        if (key <= qrow) {
            const float* kp = &Kc[((size_t)(b * S + key)) * D + h * HD];
            s = 0.0f;
            #pragma unroll 8
            for (int d = 0; d < HD; ++d) s += qb[wave][d] * kp[d];
            s *= scale;
        } else s = -INFINITY;
        sc[wave][key] = s;
        lmax = fmaxf(lmax, s);
    }
    #pragma unroll
    for (int off = 16; off; off >>= 1) lmax = fmaxf(lmax, __shfl_xor(lmax, off, 32));

    float lsum = 0.0f;
    for (int key = lane; key < S; key += 32) {
        const float p = (key <= qrow) ? __expf(sc[wave][key] - lmax) : 0.0f;
        sc[wave][key] = p;
        lsum += p;
    }
    #pragma unroll
    for (int off = 16; off; off >>= 1) lsum += __shfl_xor(lsum, off, 32);
    const float inv = 1.0f / lsum;

    float acc0 = 0.0f, acc1 = 0.0f;
    for (int key = 0; key <= qrow; ++key) {
        const float p = sc[wave][key];
        const float* vp = &Vc[((size_t)(b * S + key)) * D + h * HD];
        acc0 += p * vp[lane];
        acc1 += p * vp[lane + 32];
    }
    O[qoff + lane]      = f2bf(acc0 * inv);
    O[qoff + lane + 32] = f2bf(acc1 * inv);
}

// ---------------------------------------------------------------------------
// LayerNorm (f32 in -> bf16 out), one block per row.
// ---------------------------------------------------------------------------
__global__ __launch_bounds__(256)
void layernorm_kernel(const float* __restrict__ X, const float* __restrict__ w,
                      const float* __restrict__ bia, unsigned short* __restrict__ Y,
                      int Dd)
{
    __shared__ float red[256];
    const int row = blockIdx.x;
    const int tid = threadIdx.x;
    const float* x = X + (size_t)row * Dd;

    float s = 0.0f;
    for (int i = tid; i < Dd; i += 256) s += x[i];
    red[tid] = s; __syncthreads();
    #pragma unroll
    for (int off = 128; off; off >>= 1) {
        if (tid < off) red[tid] += red[tid + off];
        __syncthreads();
    }
    const float mean = red[0] / (float)Dd;
    __syncthreads();

    float v = 0.0f;
    for (int i = tid; i < Dd; i += 256) { const float d = x[i] - mean; v += d * d; }
    red[tid] = v; __syncthreads();
    #pragma unroll
    for (int off = 128; off; off >>= 1) {
        if (tid < off) red[tid] += red[tid + off];
        __syncthreads();
    }
    const float rstd = rsqrtf(red[0] / (float)Dd + 1e-5f);

    unsigned short* y = Y + (size_t)row * Dd;
    for (int i = tid; i < Dd; i += 256)
        y[i] = f2bf((x[i] - mean) * rstd * w[i] + bia[i]);
}

// ---------------------------------------------------------------------------
// Embedding: h = emb[tok] + pos (f32 residual stream); tok_emb kept as bf16.
// ---------------------------------------------------------------------------
__global__ __launch_bounds__(256)
void embed_kernel(const int* __restrict__ tokens, const float* __restrict__ emb,
                  const float* __restrict__ pos, unsigned short* __restrict__ tokemb_bf,
                  float* __restrict__ h)
{
    const int idx = blockIdx.x * 256 + threadIdx.x;   // < 2M
    const int d  = idx & 1023;
    const int bs = idx >> 10;
    const int s  = bs & 1023;
    const int tok = tokens[bs];
    const float te = emb[(size_t)tok * 1024 + d];
    tokemb_bf[idx] = f2bf(te);
    h[idx] = te + pos[s * 1024 + d];
}

// ---------------------------------------------------------------------------
// f32 -> bf16 conversion (weights); 4 elements per thread, n % 4 == 0.
// ---------------------------------------------------------------------------
__global__ __launch_bounds__(256)
void convert_bf16_kernel(const float* __restrict__ X, unsigned short* __restrict__ Y,
                         long long n)
{
    const long long i4 = ((long long)blockIdx.x * 256 + threadIdx.x) * 4;
    if (i4 >= n) return;
    const float4 f = *reinterpret_cast<const float4*>(&X[i4]);
    ushort4 o;
    o.x = f2bf(f.x); o.y = f2bf(f.y); o.z = f2bf(f.z); o.w = f2bf(f.w);
    *reinterpret_cast<ushort4*>(&Y[i4]) = o;
}

// ---------------------------------------------------------------------------
// Host launcher
// ---------------------------------------------------------------------------
extern "C" void kernel_launch(void* const* d_in, const int* in_sizes, int n_in,
                              void* d_out, int out_size, void* d_ws, size_t ws_size,
                              hipStream_t stream)
{
    (void)in_sizes; (void)n_in; (void)out_size; (void)ws_size;

    constexpr int V = 32000, D = 1024, L = 6, S = 1024, B = 2, FF = 4096;
    constexpr int M = B * S;

    const int*   tokens = (const int*)  d_in[0];
    const float* emb    = (const float*)d_in[1];
    const float* pos    = (const float*)d_in[2];
    const float* Wq     = (const float*)d_in[3];
    const float* Wk     = (const float*)d_in[4];
    const float* Wv     = (const float*)d_in[5];
    const float* Wo     = (const float*)d_in[6];
    const float* W1     = (const float*)d_in[7];
    const float* W2     = (const float*)d_in[8];
    const float* ln1_w  = (const float*)d_in[9];
    const float* ln1_b  = (const float*)d_in[10];
    const float* ln2_w  = (const float*)d_in[11];
    const float* ln2_b  = (const float*)d_in[12];
    const float* lnf_w  = (const float*)d_in[13];
    const float* lnf_b  = (const float*)d_in[14];
    float* logits = (float*)d_out;

    // ---- workspace layout (byte offsets, 80 MB total) ----
    char* ws = (char*)d_ws;
    const size_t MB = 1024 * 1024;
    float*          h         = (float*)         (ws + 0 * MB);   //  8 MB f32
    unsigned short* hn_bf     = (unsigned short*)(ws + 8 * MB);   //  4 MB bf16
    unsigned short* tokemb_bf = (unsigned short*)(ws + 12 * MB);  //  4 MB bf16
    float*          q         = (float*)         (ws + 16 * MB);  //  8 MB f32
    float*          kb        = (float*)         (ws + 24 * MB);  //  8 MB f32
    float*          vb        = (float*)         (ws + 32 * MB);  //  8 MB f32
    unsigned short* ao_bf     = (unsigned short*)(ws + 40 * MB);  //  4 MB bf16
    unsigned short* f1_bf     = (unsigned short*)(ws + 16 * MB);  // 16 MB (reuse q,k)
    unsigned short* wq_bf     = (unsigned short*)(ws + 44 * MB);  //  2 MB
    unsigned short* wk_bf     = (unsigned short*)(ws + 46 * MB);  //  2 MB
    unsigned short* wv_bf     = (unsigned short*)(ws + 48 * MB);  //  2 MB
    unsigned short* wo_bf     = (unsigned short*)(ws + 50 * MB);  //  2 MB
    unsigned short* w1_bf     = (unsigned short*)(ws + 52 * MB);  //  8 MB
    unsigned short* w2_bf     = (unsigned short*)(ws + 60 * MB);  //  8 MB  (ends 68)
    unsigned short* emb_bf    = (unsigned short*)(ws + 16 * MB);  // 64 MB (reuse 16..80)

    const dim3 blk(256);
    const dim3 g_embed((M * D) / 256);
    const dim3 g_ln(M);
    const dim3 g_attn(B * 16 * (S / 8));
    const dim3 g_proj(D / BN,  M / BM);   // (8,16)
    const dim3 g_ff1(FF / BN,  M / BM);   // (32,16)
    const dim3 g_out(V / BN,   M / BM);   // (250,16)
    auto cgrid = [](long long n) { return dim3((unsigned)((n / 4 + 255) / 256)); };

    embed_kernel<<<g_embed, blk, 0, stream>>>(tokens, emb, pos, tokemb_bf, h);

    const long long DD = (long long)D * D, FD = (long long)FF * D;

    for (int l = 0; l < L; ++l) {
        // stage this layer's weights as bf16
        convert_bf16_kernel<<<cgrid(DD), blk, 0, stream>>>(Wq + (size_t)l * DD, wq_bf, DD);
        convert_bf16_kernel<<<cgrid(DD), blk, 0, stream>>>(Wk + (size_t)l * DD, wk_bf, DD);
        convert_bf16_kernel<<<cgrid(DD), blk, 0, stream>>>(Wv + (size_t)l * DD, wv_bf, DD);
        convert_bf16_kernel<<<cgrid(DD), blk, 0, stream>>>(Wo + (size_t)l * DD, wo_bf, DD);
        convert_bf16_kernel<<<cgrid(FD), blk, 0, stream>>>(W1 + (size_t)l * FD, w1_bf, FD);
        convert_bf16_kernel<<<cgrid(FD), blk, 0, stream>>>(W2 + (size_t)l * FD, w2_bf, FD);

        layernorm_kernel<<<g_ln, blk, 0, stream>>>(h, ln1_w + l * D, ln1_b + l * D, hn_bf, D);

        gemm_bf16_tdm<<<g_proj, blk, 0, stream>>>(hn_bf,     wq_bf, nullptr, q,  nullptr, M, D, D, 0);
        gemm_bf16_tdm<<<g_proj, blk, 0, stream>>>(hn_bf,     wk_bf, nullptr, kb, nullptr, M, D, D, 0);
        gemm_bf16_tdm<<<g_proj, blk, 0, stream>>>(tokemb_bf, wv_bf, nullptr, vb, nullptr, M, D, D, 0);

        attention_kernel<<<g_attn, blk, 0, stream>>>(q, kb, vb, ao_bf);

        // h = h + ao @ Wo^T
        gemm_bf16_tdm<<<g_proj, blk, 0, stream>>>(ao_bf, wo_bf, h, h, nullptr, M, D, D, 2);

        layernorm_kernel<<<g_ln, blk, 0, stream>>>(h, ln2_w + l * D, ln2_b + l * D, hn_bf, D);

        // f1 = gelu(hn @ W1^T)  (bf16 out)
        gemm_bf16_tdm<<<g_ff1, blk, 0, stream>>>(hn_bf, w1_bf, nullptr, nullptr, f1_bf, M, FF, D, 1);
        // h = h + f1 @ W2^T
        gemm_bf16_tdm<<<g_proj, blk, 0, stream>>>(f1_bf, w2_bf, h, h, nullptr, M, D, FF, 2);
    }

    layernorm_kernel<<<g_ln, blk, 0, stream>>>(h, lnf_w, lnf_b, hn_bf, D);
    convert_bf16_kernel<<<cgrid((long long)V * D), blk, 0, stream>>>(emb, emb_bf, (long long)V * D);
    // logits = hn @ emb^T
    gemm_bf16_tdm<<<g_out, blk, 0, stream>>>(hn_bf, emb_bf, nullptr, logits, nullptr, M, V, D, 0);
}
